// SymmetricCrossAttentionGMU_10746008175023
// MI455X (gfx1250) — matrix operand order repaired and checked by reading, not compiled, not run on hardware
//
#include <hip/hip_runtime.h>

// ---------------------------------------------------------------------------
// SymmetricCrossAttentionGMU for MI455X (gfx1250, wave32, WMMA)
// ---------------------------------------------------------------------------
typedef _Float16 half_t;
typedef _Float16 h16 __attribute__((ext_vector_type(16)));
typedef _Float16 h8  __attribute__((ext_vector_type(8)));
typedef float    f8  __attribute__((ext_vector_type(8)));

#define BATCH 8
#define CCH   256
#define HFULL 160
#define WFULL 160
#define HD    40
#define WD    40
#define NTOK  1600           // HD*WD
#define MTOT  (BATCH * NTOK) // 12800

static __device__ __forceinline__ f8 f8zero() {
  f8 z;
#pragma unroll
  for (int g = 0; g < 8; ++g) z[g] = 0.0f;
  return z;
}

static __device__ __forceinline__ f8 wmma_f16(h16 a, h16 b, f8 c) {
  return __builtin_amdgcn_wmma_f32_16x16x32_f16(false, a, false, b, (short)0, c,
                                                false, false);
}

// A-fragment 16x32 (MxK), f16. X points at (rowbase, kbase) of a row-major
// matrix with leading dim ld (in halfs). Lane L: M = L&15, half = L>>4.
// regs 0-3: K = 8*half + {0..7}; regs 4-7: K = 16 + 8*half + {0..7}.
static __device__ __forceinline__ h16 load_a_frag(const half_t* X, int ld) {
  int lane = threadIdx.x & 31;
  int m = lane & 15, hh = lane >> 4;
  const h8* p = (const h8*)(X + (size_t)m * ld + 8 * hh);
  const h8* q = (const h8*)(X + (size_t)m * ld + 16 + 8 * hh);
  h8 lo = *p, hi = *q;
  return __builtin_shufflevector(lo, hi, 0, 1, 2, 3, 4, 5, 6, 7, 8, 9, 10, 11,
                                 12, 13, 14, 15);
}

// B-fragment 32x16 (KxN), f16, where B[k][n] = S[n][k] for a row-major source
// S with leading dim ld (rows indexed by n, contiguous k).
// Lane L: N = L&15, khalf = L>>4. regs 0-3: K = 8*khalf + {0..7}; regs 4-7: K = 16 + 8*khalf + {0..7}.
static __device__ __forceinline__ h16 load_b_frag(const half_t* S, int ld) {
  int lane = threadIdx.x & 31;
  int n = lane & 15, kh = lane >> 4;
  const h8* p = (const h8*)(S + (size_t)n * ld + 8 * kh);
  const h8* q = (const h8*)(S + (size_t)n * ld + 16 + 8 * kh);
  h8 lo = *p, hi = *q;
  return __builtin_shufflevector(lo, hi, 0, 1, 2, 3, 4, 5, 6, 7, 8, 9, 10, 11,
                                 12, 13, 14, 15);
}

// Reductions over the 16-lane halves (xor strides 1,2,4,8 never cross halves).
static __device__ __forceinline__ float red_max16(float v) {
#pragma unroll
  for (int s = 1; s <= 8; s <<= 1) v = fmaxf(v, __shfl_xor(v, s, 32));
  return v;
}
static __device__ __forceinline__ float red_sum16(float v) {
#pragma unroll
  for (int s = 1; s <= 8; s <<= 1) v += __shfl_xor(v, s, 32);
  return v;
}

static __device__ __forceinline__ float sigm(float x) {
  return 1.0f / (1.0f + __expf(-x));
}

// ---------------------------------------------------------------------------
// f32 -> f16 weight conversion
// ---------------------------------------------------------------------------
__global__ void convert_f16_kernel(const float* __restrict__ src,
                                   half_t* __restrict__ dst, int n) {
  int i = blockIdx.x * blockDim.x + threadIdx.x;
  if (i < n) dst[i] = (half_t)src[i];
}

// ---------------------------------------------------------------------------
// Bilinear /4 downsample (align_corners=False, frac==0.5 -> 2x2 average) and
// transpose [B,C,H,W] f32 -> tokens [B,N,C] f16.
// ---------------------------------------------------------------------------
__global__ void downsample_kernel(const float* __restrict__ Z,
                                  half_t* __restrict__ XD) {
  int idx = blockIdx.x * blockDim.x + threadIdx.x; // (b,c,i,j), j fastest
  if (idx >= BATCH * CCH * HD * WD) return;
  int j = idx % WD;
  int i = (idx / WD) % HD;
  int c = (idx / (WD * HD)) % CCH;
  int b = idx / (WD * HD * CCH);
  int h0 = 4 * i + 1, h1 = 4 * i + 2;
  int w0 = 4 * j + 1, w1 = 4 * j + 2;
  const float* zb = Z + ((size_t)b * CCH + c) * (HFULL * WFULL);
  float v = zb[h0 * WFULL + w0] + zb[h0 * WFULL + w1] +
            zb[h1 * WFULL + w0] + zb[h1 * WFULL + w1];
  v *= 0.25f;
  XD[((size_t)b * NTOK + i * WD + j) * CCH + c] = (half_t)v;
}

// ---------------------------------------------------------------------------
// Projection GEMM: Y[m][o] = sum_c X[m][c] * W[o][c] + bias[o]
// One wave -> 16 rows x 64 cols.  transposed=1 stores V^T layout [B,C,N].
// ---------------------------------------------------------------------------
__global__ void proj_gemm_kernel(const half_t* __restrict__ X,
                                 const half_t* __restrict__ W16,
                                 const float* __restrict__ bias,
                                 half_t* __restrict__ Y, int transposed) {
  int wave = threadIdx.x >> 5;
  int lane = threadIdx.x & 31;
  int gw = blockIdx.x * 8 + wave;          // 3200 waves
  int rt = gw >> 2;                        // 800 row tiles
  int ct = gw & 3;                         // 4 col tiles of 64
  int rowbase = rt * 16;
  int colbase = ct * 64;

  f8 acc[4];
#pragma unroll
  for (int t = 0; t < 4; ++t) acc[t] = f8zero();

#pragma unroll
  for (int ck = 0; ck < 8; ++ck) {
    h16 a = load_a_frag(X + (size_t)rowbase * CCH + ck * 32, CCH);
#pragma unroll
    for (int t = 0; t < 4; ++t) {
      h16 bfr = load_b_frag(W16 + (size_t)(colbase + t * 16) * CCH + ck * 32, CCH);
      acc[t] = wmma_f16(a, bfr, acc[t]);
    }
  }

  int n = lane & 15, hh = lane >> 4;
  int b = rowbase / NTOK;        // 16-row tile never straddles batches
  int tok0 = rowbase % NTOK;
#pragma unroll
  for (int t = 0; t < 4; ++t) {
    int o = colbase + t * 16 + n;
    float bv = bias[o];
    if (!transposed) {
#pragma unroll
      for (int g = 0; g < 8; ++g) {
        int m = rowbase + g + 8 * hh;
        Y[(size_t)m * CCH + o] = (half_t)(acc[t][g] + bv);
      }
    } else {
      h8 v;
#pragma unroll
      for (int g = 0; g < 8; ++g) v[g] = (half_t)(acc[t][g] + bv);
      half_t* p = Y + (size_t)b * CCH * NTOK + (size_t)o * NTOK + tok0 + 8 * hh;
      *(h8*)p = v; // 16B aligned: tok0+8*hh is a multiple of 8 halfs
    }
  }
}

// ---------------------------------------------------------------------------
// Flash cross-attention: O = softmax(Q K^T / 16) V
// Q,K in [B,N,C] f16; VT in [B,C,N] f16; O in [B,N,C] f16.
// One wave: 16 queries, streams 50 tiles of 32 keys with online softmax.
// ---------------------------------------------------------------------------
__global__ void attn_kernel(const half_t* __restrict__ Q,
                            const half_t* __restrict__ K,
                            const half_t* __restrict__ VT,
                            half_t* __restrict__ O) {
  __shared__ alignas(16) half_t lds_p[8][16 * 32]; // 1KB per wave
  int wave = threadIdx.x >> 5;
  int lane = threadIdx.x & 31;
  int gw = blockIdx.x * 8 + wave;   // 800 waves total
  int b = gw / (NTOK / 16);
  int mt = gw % (NTOK / 16);
  int mbase = mt * 16;
  int n = lane & 15, hh = lane >> 4;

  const half_t* Qb = Q + (size_t)b * NTOK * CCH + (size_t)mbase * CCH;
  h16 aQ[8];
#pragma unroll
  for (int ck = 0; ck < 8; ++ck) aQ[ck] = load_a_frag(Qb + ck * 32, CCH);

  f8 oacc[16];
#pragma unroll
  for (int t = 0; t < 16; ++t) oacc[t] = f8zero();
  float m_run[8], l_run[8];
#pragma unroll
  for (int g = 0; g < 8; ++g) { m_run[g] = -1e30f; l_run[g] = 0.0f; }

  const half_t* Kb0 = K + (size_t)b * NTOK * CCH;
  const half_t* Vb = VT + (size_t)b * CCH * NTOK;
  half_t* lp = lds_p[wave];

  for (int kt = 0; kt < NTOK / 32; ++kt) {
    const half_t* Kb = Kb0 + (size_t)(kt * 32) * CCH;
    f8 s0 = f8zero(), s1 = f8zero();
#pragma unroll
    for (int ck = 0; ck < 8; ++ck) {
      h16 b0 = load_b_frag(Kb + ck * 32, CCH);            // keys 0..15
      h16 b1 = load_b_frag(Kb + 16 * CCH + ck * 32, CCH); // keys 16..31
      s0 = wmma_f16(aQ[ck], b0, s0);
      s1 = wmma_f16(aQ[ck], b1, s1);
    }
    float f_[8];
#pragma unroll
    for (int g = 0; g < 8; ++g) {
      s0[g] *= 0.0625f; // 1/sqrt(256)
      s1[g] *= 0.0625f;
      float tm = red_max16(fmaxf(s0[g], s1[g]));
      float mn = fmaxf(m_run[g], tm);
      f_[g] = __expf(m_run[g] - mn);
      m_run[g] = mn;
      s0[g] = __expf(s0[g] - mn);
      s1[g] = __expf(s1[g] - mn);
      float ts = red_sum16(s0[g] + s1[g]);
      l_run[g] = l_run[g] * f_[g] + ts;
    }
#pragma unroll
    for (int t = 0; t < 16; ++t)
#pragma unroll
      for (int g = 0; g < 8; ++g) oacc[t][g] *= f_[g];

    // P (16x32 f16) to LDS, re-read as A-fragment (layout transpose).
#pragma unroll
    for (int g = 0; g < 8; ++g) {
      int m = g + 8 * hh;
      lp[m * 32 + n] = (half_t)s0[g];
      lp[m * 32 + 16 + n] = (half_t)s1[g];
    }
    __asm__ volatile("s_wait_dscnt 0" ::: "memory");
    h16 aP = load_a_frag(lp, 32);

#pragma unroll
    for (int t = 0; t < 16; ++t) {
      h16 bv = load_b_frag(Vb + (size_t)(t * 16) * NTOK + kt * 32, NTOK);
      oacc[t] = wmma_f16(aP, bv, oacc[t]);
    }
  }

  float linv[8];
#pragma unroll
  for (int g = 0; g < 8; ++g) linv[g] = 1.0f / l_run[g];
  half_t* Ob = O + (size_t)b * NTOK * CCH + (size_t)mbase * CCH;
#pragma unroll
  for (int t = 0; t < 16; ++t)
#pragma unroll
    for (int g = 0; g < 8; ++g) {
      int m = g + 8 * hh;
      Ob[(size_t)m * CCH + t * 16 + n] = (half_t)(oacc[t][g] * linv[g]);
    }
}

// ---------------------------------------------------------------------------
// Fused tail: h_rgb = tanh(Wr z_r' + br); h_ir = tanh(Wi z_i' + bi);
// z = sigmoid(sigmoid(Wg [z_r';z_i'] + bg) + temb); fused = z h_rgb + (1-z) h_ir
// Output channel-major f32 [B, C, N] for the upsampler.
// ---------------------------------------------------------------------------
__global__ void fuse_gemm_kernel(const half_t* __restrict__ ZR,
                                 const half_t* __restrict__ ZI,
                                 const half_t* __restrict__ Wg16,
                                 const half_t* __restrict__ Wr16,
                                 const half_t* __restrict__ Wi16,
                                 const float* __restrict__ bg,
                                 const float* __restrict__ br,
                                 const float* __restrict__ bi,
                                 const int* __restrict__ tidx,
                                 const float* __restrict__ temb,
                                 float* __restrict__ FT) {
  int wave = threadIdx.x >> 5;
  int lane = threadIdx.x & 31;
  int gw = blockIdx.x * 8 + wave;  // 12800 waves
  int rt = gw >> 4;                // 800 row tiles
  int ct = gw & 15;                // 16 col tiles of 16
  int rowbase = rt * 16;
  int cbase = ct * 16;

  f8 aG = f8zero(), aR = f8zero(), aI = f8zero();
#pragma unroll
  for (int ck = 0; ck < 8; ++ck) {
    h16 ar = load_a_frag(ZR + (size_t)rowbase * CCH + ck * 32, CCH);
    h16 ai = load_a_frag(ZI + (size_t)rowbase * CCH + ck * 32, CCH);
    aR = wmma_f16(ar, load_b_frag(Wr16 + (size_t)cbase * CCH + ck * 32, CCH), aR);
    aI = wmma_f16(ai, load_b_frag(Wi16 + (size_t)cbase * CCH + ck * 32, CCH), aI);
    aG = wmma_f16(ar, load_b_frag(Wg16 + (size_t)cbase * 512 + ck * 32, 512), aG);
    aG = wmma_f16(ai, load_b_frag(Wg16 + (size_t)cbase * 512 + 256 + ck * 32, 512), aG);
  }

  int n = lane & 15, hh = lane >> 4;
  int c = cbase + n;
  int b = rowbase / NTOK;
  int tok0 = rowbase % NTOK;
  float te = temb[tidx[b] * CCH + c];
  float bgv = bg[c], brv = br[c], biv = bi[c];

  float val[8];
#pragma unroll
  for (int g = 0; g < 8; ++g) {
    float hr = tanhf(aR[g] + brv);
    float hi_ = tanhf(aI[g] + biv);
    float zb = sigm(aG[g] + bgv);
    float z = sigm(zb + te);
    val[g] = z * hr + (1.0f - z) * hi_;
  }
  float* p = FT + (size_t)b * CCH * NTOK + (size_t)c * NTOK + tok0 + 8 * hh;
  *(float4*)(p + 0) = make_float4(val[0], val[1], val[2], val[3]);
  *(float4*)(p + 4) = make_float4(val[4], val[5], val[6], val[7]);
}

// ---------------------------------------------------------------------------
// Bilinear x4 upsample (align_corners=False), FT [B,C,40,40] -> out [B,C,160,160]
// ---------------------------------------------------------------------------
__global__ void upsample_kernel(const float* __restrict__ FT,
                                float* __restrict__ out) {
  int idx = blockIdx.x * blockDim.x + threadIdx.x;
  if (idx >= BATCH * CCH * HFULL * WFULL) return;
  int w = idx % WFULL;
  int h = (idx / WFULL) % HFULL;
  int c = (idx / (WFULL * HFULL)) % CCH;
  int b = idx / (WFULL * HFULL * CCH);

  float sh = fmaxf(0.25f * (float)h - 0.375f, 0.0f);
  int i0 = (int)sh;
  float fh = sh - (float)i0;
  int i1 = min(i0 + 1, HD - 1);
  float sw = fmaxf(0.25f * (float)w - 0.375f, 0.0f);
  int j0 = (int)sw;
  float fw = sw - (float)j0;
  int j1 = min(j0 + 1, WD - 1);

  const float* F = FT + ((size_t)b * CCH + c) * NTOK;
  float r0 = F[i0 * WD + j0] * (1.0f - fw) + F[i0 * WD + j1] * fw;
  float r1 = F[i1 * WD + j0] * (1.0f - fw) + F[i1 * WD + j1] * fw;
  out[idx] = r0 * (1.0f - fh) + r1 * fh;
}

// ---------------------------------------------------------------------------
extern "C" void kernel_launch(void* const* d_in, const int* in_sizes, int n_in,
                              void* d_out, int out_size, void* d_ws,
                              size_t ws_size, hipStream_t stream) {
  (void)in_sizes; (void)n_in; (void)out_size; (void)ws_size;
  const float* z_rgb = (const float*)d_in[0];
  const float* z_ir = (const float*)d_in[1];
  const int* time_idx = (const int*)d_in[2];
  const float* Wq_rgb = (const float*)d_in[3];  const float* bq_rgb = (const float*)d_in[4];
  const float* Wk_ir = (const float*)d_in[5];   const float* bk_ir = (const float*)d_in[6];
  const float* Wv_ir = (const float*)d_in[7];   const float* bv_ir = (const float*)d_in[8];
  const float* Wq_ir = (const float*)d_in[9];   const float* bq_ir = (const float*)d_in[10];
  const float* Wk_rgb = (const float*)d_in[11]; const float* bk_rgb = (const float*)d_in[12];
  const float* Wv_rgb = (const float*)d_in[13]; const float* bv_rgb = (const float*)d_in[14];
  const float* Wrgb_proj = (const float*)d_in[15]; const float* brgb_proj = (const float*)d_in[16];
  const float* Wir_proj = (const float*)d_in[17];  const float* bir_proj = (const float*)d_in[18];
  const float* Wz_gate = (const float*)d_in[19];   const float* bz_gate = (const float*)d_in[20];
  const float* temb = (const float*)d_in[21];

  char* ws = (char*)d_ws;
  const size_t HB = (size_t)BATCH * NTOK * CCH * sizeof(half_t); // 6,553,600 B
  half_t* xr    = (half_t*)(ws + 0 * HB);
  half_t* xi    = (half_t*)(ws + 1 * HB);
  half_t* q_rgb = (half_t*)(ws + 2 * HB);
  half_t* k_ir  = (half_t*)(ws + 3 * HB);
  half_t* q_ir  = (half_t*)(ws + 4 * HB);
  half_t* k_rgb = (half_t*)(ws + 5 * HB);
  half_t* vT_ir = (half_t*)(ws + 6 * HB);
  half_t* vT_rgb= (half_t*)(ws + 7 * HB);
  half_t* zr_p  = (half_t*)(ws + 8 * HB);
  half_t* zi_p  = (half_t*)(ws + 9 * HB);
  float* FT = (float*)(ws + 10 * HB);                         // 13,107,200 B
  char* wbase = ws + 10 * HB + (size_t)BATCH * CCH * NTOK * sizeof(float);
  const size_t WB = (size_t)CCH * CCH * sizeof(half_t);       // 131,072 B
  half_t* wq_rgb16 = (half_t*)(wbase + 0 * WB);
  half_t* wk_ir16  = (half_t*)(wbase + 1 * WB);
  half_t* wv_ir16  = (half_t*)(wbase + 2 * WB);
  half_t* wq_ir16  = (half_t*)(wbase + 3 * WB);
  half_t* wk_rgb16 = (half_t*)(wbase + 4 * WB);
  half_t* wv_rgb16 = (half_t*)(wbase + 5 * WB);
  half_t* wr16     = (half_t*)(wbase + 6 * WB);
  half_t* wi16     = (half_t*)(wbase + 7 * WB);
  half_t* wg16     = (half_t*)(wbase + 8 * WB);               // 2*WB

  const int CW = CCH * CCH;        // 65536
  convert_f16_kernel<<<CW / 256, 256, 0, stream>>>(Wq_rgb, wq_rgb16, CW);
  convert_f16_kernel<<<CW / 256, 256, 0, stream>>>(Wk_ir, wk_ir16, CW);
  convert_f16_kernel<<<CW / 256, 256, 0, stream>>>(Wv_ir, wv_ir16, CW);
  convert_f16_kernel<<<CW / 256, 256, 0, stream>>>(Wq_ir, wq_ir16, CW);
  convert_f16_kernel<<<CW / 256, 256, 0, stream>>>(Wk_rgb, wk_rgb16, CW);
  convert_f16_kernel<<<CW / 256, 256, 0, stream>>>(Wv_rgb, wv_rgb16, CW);
  convert_f16_kernel<<<CW / 256, 256, 0, stream>>>(Wrgb_proj, wr16, CW);
  convert_f16_kernel<<<CW / 256, 256, 0, stream>>>(Wir_proj, wi16, CW);
  convert_f16_kernel<<<2 * CW / 256, 256, 0, stream>>>(Wz_gate, wg16, 2 * CW);

  const int DSTOT = BATCH * CCH * HD * WD; // 3,276,800
  downsample_kernel<<<DSTOT / 256, 256, 0, stream>>>(z_rgb, xr);
  downsample_kernel<<<DSTOT / 256, 256, 0, stream>>>(z_ir, xi);

  // 3200 waves / 8 per block = 400 blocks each
  proj_gemm_kernel<<<400, 256, 0, stream>>>(xr, wq_rgb16, bq_rgb, q_rgb, 0);
  proj_gemm_kernel<<<400, 256, 0, stream>>>(xi, wk_ir16, bk_ir, k_ir, 0);
  proj_gemm_kernel<<<400, 256, 0, stream>>>(xi, wv_ir16, bv_ir, vT_ir, 1);
  proj_gemm_kernel<<<400, 256, 0, stream>>>(xi, wq_ir16, bq_ir, q_ir, 0);
  proj_gemm_kernel<<<400, 256, 0, stream>>>(xr, wk_rgb16, bk_rgb, k_rgb, 0);
  proj_gemm_kernel<<<400, 256, 0, stream>>>(xr, wv_rgb16, bv_rgb, vT_rgb, 1);

  // 800 waves / 8 = 100 blocks per direction
  attn_kernel<<<100, 256, 0, stream>>>(q_rgb, k_ir, vT_ir, zr_p);
  attn_kernel<<<100, 256, 0, stream>>>(q_ir, k_rgb, vT_rgb, zi_p);

  // 12800 waves / 8 = 1600 blocks
  fuse_gemm_kernel<<<1600, 256, 0, stream>>>(zr_p, zi_p, wg16, wr16, wi16,
                                             bz_gate, brgb_proj, bir_proj,
                                             time_idx, temb, FT);

  const int OTOT = BATCH * CCH * HFULL * WFULL; // 52,428,800
  upsample_kernel<<<OTOT / 256, 256, 0, stream>>>(FT, (float*)d_out);
}
